// mixprop_gat_24756191494295
// MI455X (gfx1250) — compile-verified, matrix operand order checked by reading, not run on hardware
//
#include <hip/hip_runtime.h>
#include <hip/hip_bf16.h>

typedef __attribute__((ext_vector_type(16))) __bf16 v16bf;
typedef __attribute__((ext_vector_type(8)))  float  v8f;

#define B_    8
#define C_    64
#define N_    500
#define T_    12
#define H_    4
#define F_    64
#define NT_   6000      // N_*T_
#define NPAD_ 512
#define ALPHA_ 0.05f
#define LEAK_  0.2f

// CDNA5 16-bit operand register layout (ISA 7.12.2):
// fragment element j (0..15) of a lane in half h (= lane>>4) holds K-index
//   k = 8*h + j + (j>=8 ? 8 : 0)        within a 32-wide K chunk.
// => element runs j=0..7 and j=8..15 are K-contiguous 16-byte runs.
__device__ __forceinline__ int kidx(int j, int half) {
  return 8 * half + j + ((j >= 8) ? 8 : 0);
}

union fragU  { v16bf v; uint4 q[2]; };
union pack8U { __bf16 h[8]; uint4 q; };

// 32 contiguous bytes -> v16bf
__device__ __forceinline__ v16bf load_frag32(const __bf16* p) {
  fragU u;
  u.q[0] = ((const uint4*)p)[0];
  u.q[1] = ((const uint4*)p)[1];
  return u.v;
}
// two 16-byte runs -> v16bf
__device__ __forceinline__ v16bf load_frag2(const __bf16* r0, const __bf16* r1) {
  fragU u;
  u.q[0] = *(const uint4*)r0;
  u.q[1] = *(const uint4*)r1;
  return u.v;
}

// ---------------------------------------------------------------- weight pre-swizzle
// dst fragment layout: [mat][chunk][ft][lane][16] bf16, value = src[mat*ms + k*sk + o*so]
__global__ void swizzle_w_kernel(const float* __restrict__ src, int K, long sk, long so,
                                 int nmat, long ms, __bf16* __restrict__ dst) {
  int idx = blockIdx.x * blockDim.x + threadIdx.x;
  int per = K * 64;
  if (idx >= nmat * per) return;
  int mat = idx / per, r = idx % per;
  int e = r & 15, lane = (r >> 4) & 31, ft = (r >> 9) & 3, chunk = r >> 11;
  int half = lane >> 4, fl = lane & 15;
  int k = chunk * 32 + kidx(e, half);
  int o = ft * 16 + fl;
  dst[idx] = (__bf16)src[mat * ms + (long)k * sk + (long)o * so];
}

// ---------------------------------------------------------------- mask bits
__global__ void mask_kernel(const float* __restrict__ adj, unsigned* __restrict__ mb) {
  int idx = blockIdx.x * blockDim.x + threadIdx.x;       // 512*16
  if (idx >= NPAD_ * 16) return;
  int n = idx >> 4, w = idx & 15;
  unsigned bits = 0u;
  if (n < N_) {
    for (int j = 0; j < 32; ++j) {
      int m = w * 32 + j;
      if (m < N_ && (adj[n * N_ + m] > 0.f || m == n)) bits |= (1u << j);
    }
  }
  mb[idx] = bits;
}

// ---------------------------------------------------------------- x -> outs slot 0
__global__ void copy_x_kernel(const float* __restrict__ x, float* __restrict__ outs) {
  long idx = (long)blockIdx.x * blockDim.x + threadIdx.x;
  const long per_b = (long)C_ * NT_;
  if (idx >= (long)B_ * per_b) return;
  long b = idx / per_b, r = idx % per_b;
  outs[b * (3L * C_ * NT_) + r] = x[idx];
}

// ---------------------------------------------------------------- head projection
// WhSwz[(b,h,t)][chunk][ft][lane][16] (bf16) = fragment-swizzled hp @ W
// e1/e2[b,h,t,npad] = Wh . a1/a2
__global__ void proj_kernel(const float* __restrict__ h, long sb_h,
                            const __bf16* __restrict__ Wswz,
                            const float* __restrict__ a1, const float* __restrict__ a2,
                            __bf16* __restrict__ WhSwz,
                            float* __restrict__ e1, float* __restrict__ e2) {
  __shared__ __bf16 hp[16][72];                 // 144B row stride (16B multiple)
  const int ntile = blockIdx.x;                 // 0..31
  const int bt = blockIdx.y;
  const int b = bt / T_, t = bt % T_;
  const int tid = threadIdx.x;

  for (int i = tid; i < 16 * 64; i += 128) {
    int row = i >> 6, c = i & 63;
    int n = ntile * 16 + row;
    float v = (n < N_) ? h[(long)b * sb_h + (long)c * NT_ + (long)n * T_ + t] : 0.f;
    hp[row][c] = (__bf16)v;
  }
  __syncthreads();

  const int hd = tid >> 5;                      // wave == head
  const int lane = tid & 31;
  const int half = lane >> 4, mr = lane & 15;

  v8f acc[4] = {v8f{}, v8f{}, v8f{}, v8f{}};
#pragma unroll
  for (int kc = 0; kc < 2; ++kc) {
    int kk = kc * 32;
    v16bf a = load_frag2(&hp[mr][kk + 8 * half], &hp[mr][kk + 16 + 8 * half]);
#pragma unroll
    for (int ft = 0; ft < 4; ++ft) {
      v16bf bb = load_frag32(Wswz + ((long)hd * 4096) + ((kc * 4 + ft) * 32 + lane) * 16);
      acc[ft] = __builtin_amdgcn_wmma_f32_16x16x32_bf16(
          false, a, false, bb, (short)0, acc[ft], false, false);
    }
  }

  // --- store Wh in B-fragment-swizzled order: lane-local, contiguous 16B per ft
  const long whbase = ((long)(b * H_ + hd) * T_ + t) * ((long)16 * 4 * 32 * 16);
  const int  chunk  = ntile >> 1;
  const int  joff   = (ntile & 1) * 8;
#pragma unroll
  for (int ft = 0; ft < 4; ++ft) {
    pack8U p;
#pragma unroll
    for (int v = 0; v < 8; ++v) p.h[v] = (__bf16)acc[ft][v];
    *(uint4*)(WhSwz + whbase + (long)((chunk * 4 + ft) * 32 + lane) * 16 + joff) = p.q;
  }

  // --- e1/e2 (rows of C layout: lane holds rows v + 8*half, col ft*16+mr)
  float p1[8] = {}, p2[8] = {};
#pragma unroll
  for (int ft = 0; ft < 4; ++ft) {
    int f = ft * 16 + mr;
    float A1 = a1[hd * 64 + f], A2 = a2[hd * 64 + f];
#pragma unroll
    for (int v = 0; v < 8; ++v) {
      p1[v] += acc[ft][v] * A1;
      p2[v] += acc[ft][v] * A2;
    }
  }
#pragma unroll
  for (int v = 0; v < 8; ++v) {
#pragma unroll
    for (int off = 1; off < 16; off <<= 1) {
      p1[v] += __shfl_xor(p1[v], off, 32);
      p2[v] += __shfl_xor(p2[v], off, 32);
    }
  }
  if (mr == 0) {
    long eb = ((long)(b * H_ + hd) * T_ + t) * NPAD_ + ntile * 16 + 8 * half;
#pragma unroll
    for (int v = 0; v < 8; ++v) { e1[eb + v] = p1[v]; e2[eb + v] = p2[v]; }
  }
}

// ---------------------------------------------------------------- fused attention
// scores -> masked leaky-relu -> softmax (2-pass) -> P@Wh (WMMA) -> ELU -> cat
__global__ void att_kernel(const float* __restrict__ e1g, const float* __restrict__ e2g,
                           const unsigned* __restrict__ mb,
                           const __bf16* __restrict__ WhSwz, float* __restrict__ cat) {
  __shared__ float e2s[NPAD_];
  const int rblk = blockIdx.x;                  // 0..7 (64 rows each)
  const int bt = blockIdx.y;
  const int b = bt / T_, t = bt % T_;
  const int hd = blockIdx.z;
  const int tid = threadIdx.x, wv = tid >> 5, lane = tid & 31;
  const int half = lane >> 4, mr = lane & 15;

  const long ebase = ((long)(b * H_ + hd) * T_ + t) * NPAD_;
  const __bf16* wh = WhSwz + ebase * 64;        // [chunk][ft][lane][16]

  for (int i = tid; i < NPAD_; i += 128) e2s[i] = e2g[ebase + i];

  const int row = rblk * 64 + wv * 16 + mr;     // 0..511
  const float E1 = e1g[ebase + row];
  unsigned mw[16];
#pragma unroll
  for (int i = 0; i < 16; ++i) mw[i] = mb[(long)row * 16 + i];
  __syncthreads();

  // pass A: row max of masked leaky-relu scores
  float vmax = -3.0e38f;
  for (int mc = 0; mc < NPAD_; mc += 32) {
    unsigned w = mw[mc >> 5];
    float4 r0 = *(const float4*)&e2s[mc + 8 * half];
    float4 r1 = *(const float4*)&e2s[mc + 8 * half + 4];
    float4 r2 = *(const float4*)&e2s[mc + 16 + 8 * half];
    float4 r3 = *(const float4*)&e2s[mc + 16 + 8 * half + 4];
    float ev[16] = {r0.x, r0.y, r0.z, r0.w, r1.x, r1.y, r1.z, r1.w,
                    r2.x, r2.y, r2.z, r2.w, r3.x, r3.y, r3.z, r3.w};
#pragma unroll
    for (int j = 0; j < 16; ++j) {
      int k = kidx(j, half);
      float sv = E1 + ev[j];
      sv = sv > 0.f ? sv : LEAK_ * sv;
      sv = ((w >> k) & 1u) ? sv : -9.0e15f;
      vmax = fmaxf(vmax, sv);
    }
  }
  vmax = fmaxf(vmax, __shfl_xor(vmax, 16, 32));

  // pass B: p = exp(s - max); acc += P @ Wh   (B frags: coalesced 32B/lane loads)
  v8f acc[4] = {v8f{}, v8f{}, v8f{}, v8f{}};
  float lsum = 0.f;
  for (int mc = 0; mc < NPAD_; mc += 32) {
    int chunk = mc >> 5;
    unsigned w = mw[chunk];
    float4 r0 = *(const float4*)&e2s[mc + 8 * half];
    float4 r1 = *(const float4*)&e2s[mc + 8 * half + 4];
    float4 r2 = *(const float4*)&e2s[mc + 16 + 8 * half];
    float4 r3 = *(const float4*)&e2s[mc + 16 + 8 * half + 4];
    float ev[16] = {r0.x, r0.y, r0.z, r0.w, r1.x, r1.y, r1.z, r1.w,
                    r2.x, r2.y, r2.z, r2.w, r3.x, r3.y, r3.z, r3.w};
    v16bf a;
#pragma unroll
    for (int j = 0; j < 16; ++j) {
      int k = kidx(j, half);
      float sv = E1 + ev[j];
      sv = sv > 0.f ? sv : LEAK_ * sv;
      sv = ((w >> k) & 1u) ? sv : -9.0e15f;
      float p = __expf(sv - vmax);
      lsum += p;
      a[j] = (__bf16)p;
    }
#pragma unroll
    for (int ft = 0; ft < 4; ++ft) {
      v16bf bb = load_frag32(wh + (long)((chunk * 4 + ft) * 32 + lane) * 16);
      acc[ft] = __builtin_amdgcn_wmma_f32_16x16x32_bf16(
          false, a, false, bb, (short)0, acc[ft], false, false);
    }
  }
  lsum += __shfl_xor(lsum, 16, 32);
  float inv = 1.0f / lsum;

  const int n0 = rblk * 64 + wv * 16;
#pragma unroll
  for (int ft = 0; ft < 4; ++ft) {
    int f = ft * 16 + mr;
#pragma unroll
    for (int v = 0; v < 8; ++v) {
      int rr = v + 8 * half;
      float sc = __shfl(inv, rr, 32);
      float val = acc[ft][v] * sc;
      val = val > 0.f ? val : (__expf(val) - 1.f);  // ELU
      int n = n0 + rr;
      if (n < N_)
        cat[(((long)b * (H_ * F_) + hd * F_ + f) * N_ + n) * T_ + t] = val;
    }
  }
}

// ---------------------------------------------------------------- generic WMMA 1x1 conv
// dst[b,o,r] = (xres? a*x + (1-a)*.. : ..)(sum_k src[b,k,r]*W[o,k] + bias[o])
__global__ void conv1x1_kernel(const float* __restrict__ src, long sb_src, int K,
                               const __bf16* __restrict__ Wswz, const float* __restrict__ bias,
                               const float* __restrict__ xres, long sb_x, float alpha,
                               float* __restrict__ dst, long sb_dst) {
  extern __shared__ __bf16 sa[];                // [128][K+8]
  const int b = blockIdx.y;
  const int r0 = blockIdx.x * 128;
  const int tid = threadIdx.x, wv = tid >> 5, lane = tid & 31;
  const int half = lane >> 4, mr = lane & 15;
  const int KS = K + 8;

  for (int i = tid; i < 128 * K; i += 256) {
    int k = i / 128, row = i % 128;             // coalesced across rows
    int rg = r0 + row;
    float v = (rg < NT_) ? src[(long)b * sb_src + (long)k * NT_ + rg] : 0.f;
    sa[(long)row * KS + k] = (__bf16)v;
  }
  __syncthreads();

  const __bf16* arow = sa + (long)(wv * 16 + mr) * KS;
  v8f acc[4] = {v8f{}, v8f{}, v8f{}, v8f{}};
  for (int kk = 0; kk < K; kk += 32) {
    v16bf a = load_frag2(arow + kk + 8 * half, arow + kk + 16 + 8 * half);
    int chunk = kk >> 5;
#pragma unroll
    for (int ft = 0; ft < 4; ++ft) {
      v16bf bb = load_frag32(Wswz + (long)((chunk * 4 + ft) * 32 + lane) * 16);
      acc[ft] = __builtin_amdgcn_wmma_f32_16x16x32_bf16(
          false, a, false, bb, (short)0, acc[ft], false, false);
    }
  }

#pragma unroll
  for (int ft = 0; ft < 4; ++ft) {
    int o = ft * 16 + mr;
    float bv = bias[o];
#pragma unroll
    for (int v = 0; v < 8; ++v) {
      int rg = r0 + wv * 16 + v + 8 * half;
      if (rg < NT_) {
        float val = acc[ft][v] + bv;
        if (xres) {
          float xv = xres[(long)b * sb_x + (long)o * NT_ + rg];
          val = alpha * xv + (1.f - alpha) * val;
        }
        dst[(long)b * sb_dst + (long)o * NT_ + rg] = val;
      }
    }
  }
}

// ---------------------------------------------------------------- launch
extern "C" void kernel_launch(void* const* d_in, const int* in_sizes, int n_in,
                              void* d_out, int out_size, void* d_ws, size_t ws_size,
                              hipStream_t stream) {
  const float* x   = (const float*)d_in[0];
  const float* adj = (const float*)d_in[1];
  const float* W   = (const float*)d_in[2];
  const float* a1  = (const float*)d_in[3];
  const float* a2  = (const float*)d_in[4];
  const float* Wg  = (const float*)d_in[5];
  const float* bg  = (const float*)d_in[6];
  const float* Wm  = (const float*)d_in[7];
  const float* bm  = (const float*)d_in[8];
  float* out = (float*)d_out;

  char* ws = (char*)d_ws;
  auto carve = [&](size_t bytes) -> char* {
    char* p = ws; ws += (bytes + 255) & ~(size_t)255; return p;
  };
  unsigned* maskbits = (unsigned*)carve((size_t)NPAD_ * 16 * 4);
  float*    e1       = (float*)   carve((size_t)B_ * H_ * T_ * NPAD_ * 4);
  float*    e2       = (float*)   carve((size_t)B_ * H_ * T_ * NPAD_ * 4);
  __bf16*   WhSwz    = (__bf16*)  carve((size_t)B_ * H_ * T_ * NPAD_ * 64 * 2);
  float*    cat      = (float*)   carve((size_t)B_ * (H_ * F_) * NT_ * 4);
  float*    outs     = (float*)   carve((size_t)B_ * 3 * C_ * NT_ * 4);
  __bf16*   Wswz     = (__bf16*)  carve((size_t)H_ * 64 * 64 * 2);
  __bf16*   Wgswz    = (__bf16*)  carve((size_t)256 * 64 * 2);
  __bf16*   Wmswz    = (__bf16*)  carve((size_t)192 * 64 * 2);

  // weight pre-swizzle (once per launch; tiny)
  swizzle_w_kernel<<<(H_ * 64 * 64 + 255) / 256, 256, 0, stream>>>(
      W, 64, 64, 1, H_, 4096, Wswz);                    // W[h][c][f]: k=c, o=f
  swizzle_w_kernel<<<(256 * 64 + 255) / 256, 256, 0, stream>>>(
      Wg, 256, 1, 256, 1, 0, Wgswz);                    // Wg[o][k]
  swizzle_w_kernel<<<(192 * 64 + 255) / 256, 256, 0, stream>>>(
      Wm, 192, 1, 192, 1, 0, Wmswz);                    // Wm[o][k]

  mask_kernel<<<(NPAD_ * 16 + 255) / 256, 256, 0, stream>>>(adj, maskbits);
  copy_x_kernel<<<((long)B_ * C_ * NT_ + 255) / 256, 256, 0, stream>>>(x, outs);

  for (int g = 0; g < 2; ++g) {
    const float* hsrc = (g == 0) ? x : (outs + (long)g * C_ * NT_);
    long sbh = (g == 0) ? (long)C_ * NT_ : (long)3 * C_ * NT_;
    proj_kernel<<<dim3(32, B_ * T_), 128, 0, stream>>>(hsrc, sbh, Wswz, a1, a2,
                                                       WhSwz, e1, e2);
    att_kernel<<<dim3(8, B_ * T_, H_), 128, 0, stream>>>(e1, e2, maskbits, WhSwz, cat);
    conv1x1_kernel<<<dim3((NT_ + 127) / 128, B_), 256, 128 * (256 + 8) * 2, stream>>>(
        cat, (long)(H_ * F_) * NT_, 256, Wgswz, bg,
        x, (long)C_ * NT_, ALPHA_,
        outs + (long)(g + 1) * C_ * NT_, (long)3 * C_ * NT_);
  }
  conv1x1_kernel<<<dim3((NT_ + 127) / 128, B_), 256, 128 * (192 + 8) * 2, stream>>>(
      outs, (long)3 * C_ * NT_, 192, Wmswz, bm,
      nullptr, 0, 0.f,
      out, (long)C_ * NT_);
}
